// NCP_Model_26276609917072
// MI455X (gfx1250) — matrix operand order repaired and checked by reading, not compiled, not run on hardware
//
#include <hip/hip_runtime.h>
#include <hip/hip_bf16.h>

typedef _Float16 v8h  __attribute__((ext_vector_type(8)));
typedef _Float16 v16h __attribute__((ext_vector_type(16)));
typedef float    v8f  __attribute__((ext_vector_type(8)));

__device__ __forceinline__ v16h cat8(v8h lo, v8h hi) {
    return __builtin_shufflevector(lo, hi, 0,1,2,3,4,5,6,7,8,9,10,11,12,13,14,15);
}

__device__ __forceinline__ v8f wmma_f16(v16h a, v16h b, v8f c) {
    return __builtin_amdgcn_wmma_f32_16x16x32_f16(false, a, false, b, (short)0, c,
                                                  false, false);
}

// ---------------------------------------------------------------------------
// Pack f32 weights [Cout][Kreal] into zero-padded f16 [CoutPad][Kpad].
// ---------------------------------------------------------------------------
__global__ void pack_weights_f16(const float* __restrict__ src, _Float16* __restrict__ dst,
                                 int Cout, int Kreal, int Kpad, int total) {
    int idx = blockIdx.x * 256 + threadIdx.x;
    if (idx >= total) return;
    int m = idx / Kpad;
    int k = idx - m * Kpad;
    float v = (m < Cout && k < Kreal) ? src[m * Kreal + k] : 0.0f;
    dst[idx] = (_Float16)v;
}

// ---------------------------------------------------------------------------
// Implicit-GEMM convolution with WMMA. All shape parameters are compile-time:
// divisions fold to shifts/mads, out-of-range guards constant-fold.
// Block: 256 threads (8 wave32). Block tile: 16 (Cout) x 256 (pixels).
// Each wave owns a 16x32 strip -> 2 v_wmma per 32-wide K step, A reused.
// LDS rows padded to 40 halves (80B): 20-bank stride, conflict-free frag reads.
// ---------------------------------------------------------------------------
template <int Cin, int H, int W, int Cout, int KH, int KW, int STR,
          int OH, int OW, int Kreal, int Kpad>
__global__ void __launch_bounds__(256)
conv_wmma(const float* __restrict__ in, const _Float16* __restrict__ Wt,
          const float* __restrict__ bias, float* __restrict__ out) {
    constexpr int OHW  = OH * OW;
    constexpr int KHKW = KH * KW;
    constexpr int ROWP = 40;

    __shared__ __align__(16) _Float16 As[16][ROWP];   // [m][k]
    __shared__ __align__(16) _Float16 Bs[256][ROWP];  // [pixel][k]

    const int tid   = threadIdx.x;
    const int lane  = tid & 31;
    const int wave  = tid >> 5;
    const int mtile = blockIdx.x;
    const int img   = blockIdx.z;
    const int pixBase = blockIdx.y * 256;

    // This thread stages im2col row `tid` (pixel pixBase+tid), all 32 K values.
    const int pFill = pixBase + tid;
    const bool pValid = (pFill < OHW);
    const int oy = pValid ? (pFill / OW) : 0;
    const int ox = pValid ? (pFill - oy * OW) : 0;
    const float* __restrict__ pIn =
        in + (long)img * (Cin * H * W) + (oy * STR) * W + (ox * STR);

    v8f acc0 = {}, acc1 = {};

    #pragma unroll 1
    for (int k0 = 0; k0 < Kpad; k0 += 32) {
        // --- stage A (weights): 512 halves, one 32-bit copy per thread ---
        {
            int idx = tid * 2;
            int m  = idx >> 5;
            int kk = idx & 31;
            const unsigned int* s =
                (const unsigned int*)(Wt + (mtile * 16 + m) * Kpad + k0 + kk);
            *(unsigned int*)&As[m][kk] = *s;
        }
        // --- stage B (im2col gather): incremental (cin,kh,kw) walk ---
        {
            int cin = k0 / KHKW;
            int r   = k0 - cin * KHKW;
            int kh  = r / KW;
            int kw  = r - kh * KW;
            #pragma unroll
            for (int kk = 0; kk < 32; ++kk) {
                float v = 0.0f;
                if (pValid && (k0 + kk) < Kreal)
                    v = pIn[(cin * H + kh) * W + kw];
                Bs[tid][kk] = (_Float16)v;
                if (++kw == KW) { kw = 0; if (++kh == KH) { kh = 0; ++cin; } }
            }
            // prefetch next K-step's first input element (global_prefetch_b8)
            if (pValid && (k0 + 32) < Kreal) {
                int kn = k0 + 32;
                int cn = kn / KHKW, rn = kn - cn * KHKW;
                int hn = rn / KW,  wn = rn - hn * KW;
                __builtin_prefetch(pIn + (cn * H + hn) * W + wn, 0, 0);
            }
        }
        __syncthreads();

        // --- fragments in contiguous k order (k = 16*hi + s) for A and B; 2 WMMAs, A reused ---
        const int hi = lane >> 4;
        const int ml = lane & 15;
        const v8h* arow = (const v8h*)(&As[ml][0]);
        v16h afrag = cat8(arow[hi * 2], arow[hi * 2 + 1]);
        const v8h* brow0 = (const v8h*)(&Bs[wave * 32 + ml][0]);
        v16h b0 = cat8(brow0[hi * 2], brow0[hi * 2 + 1]);
        const v8h* brow1 = (const v8h*)(&Bs[wave * 32 + 16 + ml][0]);
        v16h b1 = cat8(brow1[hi * 2], brow1[hi * 2 + 1]);
        acc0 = wmma_f16(afrag, b0, acc0);
        acc1 = wmma_f16(afrag, b1, acc1);
        __syncthreads();
    }

    // --- store: lane -> N, VGPR r -> M (+8 for upper half-lanes); bias+ReLU ---
    const int mB = (lane >> 4) * 8;
    #pragma unroll
    for (int sub = 0; sub < 2; ++sub) {
        int n = pixBase + wave * 32 + sub * 16 + (lane & 15);
        v8f& acc = sub ? acc1 : acc0;
        if (n < OHW) {
            #pragma unroll
            for (int r = 0; r < 8; ++r) {
                int co = mtile * 16 + mB + r;
                if (co < Cout) {
                    float v = acc[r] + bias[co];
                    out[((long)img * Cout + co) * OHW + n] = v > 0.0f ? v : 0.0f;
                }
            }
        }
    }
}

// ---------------------------------------------------------------------------
// FC layer: out[512,128] = relu(A[512,1152](f32) x B[1152,128](f16) + bias).
// K staged 64 wide per barrier pair -> 2 WMMAs per step.
// ---------------------------------------------------------------------------
__global__ void __launch_bounds__(256)
gemm_fc_wmma(const float* __restrict__ A, const _Float16* __restrict__ Bm,
             const float* __restrict__ bias, float* __restrict__ out) {
    constexpr int N = 128, K = 1152;
    constexpr int ROWP = 72;   // 144B rows: 36-bank stride, conflict-free

    __shared__ __align__(16) _Float16 As[16][ROWP];
    __shared__ __align__(16) _Float16 Bs[128][ROWP];

    const int tid   = threadIdx.x;
    const int lane  = tid & 31;
    const int wave  = tid >> 5;
    const int mtile = blockIdx.x;

    v8f acc = {};

    #pragma unroll 1
    for (int k0 = 0; k0 < K; k0 += 64) {
        {   // A tile: 16x64 f32->f16, 4 contiguous per thread (b128 load)
            int idx = tid * 4;
            int m  = idx >> 6;
            int kk = idx & 63;
            const float* s = A + (mtile * 16 + m) * K + k0 + kk;
            As[m][kk]     = (_Float16)s[0];
            As[m][kk + 1] = (_Float16)s[1];
            As[m][kk + 2] = (_Float16)s[2];
            As[m][kk + 3] = (_Float16)s[3];
        }
        {   // B tile: 128x64, thread -> (row tid&127, K half (tid>>7)*32)
            int n  = tid & 127;
            int kb = (tid >> 7) * 32;
            #pragma unroll
            for (int u = 0; u < 32; ++u)
                Bs[n][kb + u] = Bm[(k0 + kb + u) * N + n];
        }
        __syncthreads();

        const int hi = lane >> 4;
        const int ml = lane & 15;
        const v8h* arow = (const v8h*)(&As[ml][0]);
        v16h a0 = cat8(arow[hi * 2],     arow[hi * 2 + 1]);   // K 0..31
        v16h a1 = cat8(arow[4 + hi * 2], arow[4 + hi * 2 + 1]);   // K 32..63
        const v8h* brow = (const v8h*)(&Bs[wave * 16 + ml][0]);
        v16h b0 = cat8(brow[hi * 2],     brow[hi * 2 + 1]);
        v16h b1 = cat8(brow[4 + hi * 2], brow[4 + hi * 2 + 1]);
        acc = wmma_f16(a0, b0, acc);
        acc = wmma_f16(a1, b1, acc);
        __syncthreads();
    }

    const int n  = wave * 16 + (lane & 15);
    const int mB = (lane >> 4) * 8;
    #pragma unroll
    for (int r = 0; r < 8; ++r) {
        int row = mtile * 16 + mB + r;
        float v = acc[r] + bias[n];
        out[(long)row * N + n] = v > 0.0f ? v : 0.0f;
    }
}

// ---------------------------------------------------------------------------
// LTC recurrence. One block per batch element, 64 threads = 64 neurons.
// ---------------------------------------------------------------------------
__global__ void __launch_bounds__(64)
ltc_kernel(const float* __restrict__ feats,
           const float* __restrict__ s_mu, const float* __restrict__ s_sigma,
           const float* __restrict__ s_w,  const float* __restrict__ s_erev,
           const float* __restrict__ mu,   const float* __restrict__ sigma,
           const float* __restrict__ wmat, const float* __restrict__ erev,
           const float* __restrict__ gleak, const float* __restrict__ vleak,
           const float* __restrict__ cm,   float* __restrict__ out, int T) {
    const int b = blockIdx.x;
    const int j = threadIdx.x;
    __shared__ float inp[128];
    __shared__ float vsh[64];

    const float gl  = gleak[j];
    const float glv = gl * vleak[j];
    const float cmt = cm[j] * 6.0f;
    float v = 0.0f;
    vsh[j] = 0.0f;
    __syncthreads();

    for (int t = 0; t < T; ++t) {
        const float* f = feats + (long)(b * T + t) * 128;
        inp[j]      = f[j];
        inp[j + 64] = f[j + 64];
        __syncthreads();

        float numS = 0.0f, denS = 0.0f;
        for (int i = 0; i < 128; ++i) {
            int idx = i * 64 + j;
            float x   = (inp[i] - s_mu[idx]) * s_sigma[idx];
            float act = s_w[idx] / (1.0f + __expf(-x));
            numS += act * s_erev[idx];
            denS += act;
        }

        for (int u = 0; u < 6; ++u) {
            float num = numS, den = denS;
            for (int i = 0; i < 64; ++i) {
                int idx = i * 64 + j;
                float x   = (vsh[i] - mu[idx]) * sigma[idx];
                float act = wmat[idx] / (1.0f + __expf(-x));
                num += act * erev[idx];
                den += act;
            }
            v = (cmt * v + glv + num) / (cmt + gl + den + 1e-8f);
            __syncthreads();
            vsh[j] = v;
            __syncthreads();
        }
        if (j < 3) out[(long)(b * T + t) * 3 + j] = v;
        __syncthreads();
    }
}

// ---------------------------------------------------------------------------
// Host-side orchestration
// ---------------------------------------------------------------------------
extern "C" void kernel_launch(void* const* d_in, const int* in_sizes, int n_in,
                              void* d_out, int out_size, void* d_ws, size_t ws_size,
                              hipStream_t stream) {
    (void)in_sizes; (void)n_in; (void)out_size; (void)ws_size;

    const float* x     = (const float*)d_in[0];
    const float* wc[5] = {(const float*)d_in[1], (const float*)d_in[3],
                          (const float*)d_in[5], (const float*)d_in[7],
                          (const float*)d_in[9]};
    const float* bc[5] = {(const float*)d_in[2], (const float*)d_in[4],
                          (const float*)d_in[6], (const float*)d_in[8],
                          (const float*)d_in[10]};
    const float* fc_w  = (const float*)d_in[11];
    const float* fc_b  = (const float*)d_in[12];
    float* out = (float*)d_out;

    const int BT = 32 * 16;
    // Cout, outHW, Kreal, Kpad, CoutPad per layer
    const int Cout[5]    = {24, 36, 48, 64, 64};
    const int OutHW[5]   = {31 * 98, 14 * 47, 5 * 22, 3 * 20, 1 * 18};
    const int Kreal[5]   = {75, 600, 900, 432, 576};
    const int Kpad[5]    = {96, 608, 928, 448, 576};
    const int CoutPad[5] = {32, 48, 48, 64, 64};

    char* base = (char*)d_ws;
    size_t off = 0;
    auto carve = [&](size_t bytes) -> char* {
        char* p = base + off;
        off = (off + bytes + 255) & ~(size_t)255;
        return p;
    };
    _Float16* wf[5];
    for (int i = 0; i < 5; ++i)
        wf[i] = (_Float16*)carve((size_t)CoutPad[i] * Kpad[i] * sizeof(_Float16));
    _Float16* fcf = (_Float16*)carve((size_t)1152 * 128 * sizeof(_Float16));
    float* act[5];
    for (int i = 0; i < 5; ++i)
        act[i] = (float*)carve((size_t)BT * Cout[i] * OutHW[i] * sizeof(float));
    float* feat = (float*)carve((size_t)BT * 128 * sizeof(float));

    // 1) Pack weights to padded f16
    for (int i = 0; i < 5; ++i) {
        int total = CoutPad[i] * Kpad[i];
        pack_weights_f16<<<(total + 255) / 256, 256, 0, stream>>>(
            wc[i], wf[i], Cout[i], Kreal[i], Kpad[i], total);
    }
    {
        int total = 1152 * 128;
        pack_weights_f16<<<(total + 255) / 256, 256, 0, stream>>>(
            fc_w, fcf, 1152, 128, 128, total);
    }

    // 2) Conv stack (compile-time shapes -> implicit-GEMM WMMA)
    auto gy = [](int ohw) { return (ohw + 255) / 256; };
    conv_wmma< 3, 66, 200, 24, 5, 5, 2, 31, 98,  75,  96>
        <<<dim3(2, gy(OutHW[0]), BT), 256, 0, stream>>>(x,      wf[0], bc[0], act[0]);
    conv_wmma<24, 31,  98, 36, 5, 5, 2, 14, 47, 600, 608>
        <<<dim3(3, gy(OutHW[1]), BT), 256, 0, stream>>>(act[0], wf[1], bc[1], act[1]);
    conv_wmma<36, 14,  47, 48, 5, 5, 2,  5, 22, 900, 928>
        <<<dim3(3, gy(OutHW[2]), BT), 256, 0, stream>>>(act[1], wf[2], bc[2], act[2]);
    conv_wmma<48,  5,  22, 64, 3, 3, 1,  3, 20, 432, 448>
        <<<dim3(4, gy(OutHW[3]), BT), 256, 0, stream>>>(act[2], wf[3], bc[3], act[3]);
    conv_wmma<64,  3,  20, 64, 3, 3, 1,  1, 18, 576, 576>
        <<<dim3(4, gy(OutHW[4]), BT), 256, 0, stream>>>(act[3], wf[4], bc[4], act[4]);

    // 3) FC: [512,1152] x [1152,128] + bias, ReLU
    gemm_fc_wmma<<<dim3(BT / 16, 1, 1), 256, 0, stream>>>(act[4], fcf, fc_b, feat);

    // 4) LTC recurrence over T
    ltc_kernel<<<32, 64, 0, stream>>>(feat,
        (const float*)d_in[13], (const float*)d_in[14], (const float*)d_in[15],
        (const float*)d_in[16], (const float*)d_in[17], (const float*)d_in[18],
        (const float*)d_in[19], (const float*)d_in[20], (const float*)d_in[21],
        (const float*)d_in[22], (const float*)d_in[23], out, 16);
}
